// QuantizedLinear_86337432584511
// MI455X (gfx1250) — matrix operand order repaired and checked by reading, not compiled, not run on hardware
//
#include <hip/hip_runtime.h>
#include <hip/hip_bf16.h>

// ---- sizes (compile-time, from the reference) ----
#define TOK_M 4096      // 2*2048 tokens
#define OUT_N 11008
#define IN_K  4096
#define QCOLS 2048      // weight_q columns (1 byte -> 2 nf4 values)

// ---- plain ext-vector types (union-safe) ----
typedef __attribute__((ext_vector_type(16))) __bf16 v16bf;
typedef __attribute__((ext_vector_type(8)))  float  v8f;
typedef __attribute__((ext_vector_type(4)))  int    i32x4;
typedef __attribute__((ext_vector_type(4)))  float  f32x4;
typedef __attribute__((ext_vector_type(2)))  unsigned int u32x2;

__device__ __constant__ float NF4_LUT[16] = {
    -1.0f, -0.6961928f, -0.525073f, -0.39491748f, -0.28444138f,
    -0.18477343f, -0.09105602f, 0.0f, 0.07958029f, 0.1609302f,
    0.2461123f, 0.33791524f, 0.44070983f, 0.56261897f,
    0.72295684f, 1.0f};

__device__ __forceinline__ unsigned int f32_to_bf16_bits(float f) {
    unsigned int u = __float_as_uint(f);
    u += 0x7FFFu + ((u >> 16) & 1u);   // round-to-nearest-even
    return u >> 16;
}

// ---------------------------------------------------------------------------
// Kernel 1: NF4 dequant  Wq(int32 "bytes") -> bf16 W (N x K row-major)
// ---------------------------------------------------------------------------
__global__ __launch_bounds__(256)
void nf4_dequant_kernel(const int* __restrict__ wq,
                        const float* __restrict__ scale,
                        unsigned int* __restrict__ wb) {
    const int idx = blockIdx.x * 256 + threadIdx.x;   // 0 .. N*QCOLS-1
    const int o = idx >> 11;                          // / 2048
    const int b = wq[idx];
    const float s = scale[o];
    const float hi = NF4_LUT[(b >> 4) & 15] * s;      // element 2j
    const float lo = NF4_LUT[b & 15] * s;             // element 2j+1
    wb[idx] = f32_to_bf16_bits(hi) | (f32_to_bf16_bits(lo) << 16);
}

// ---------------------------------------------------------------------------
// Kernel 2: X f32 -> bf16 (4 elements / thread)
// ---------------------------------------------------------------------------
__global__ __launch_bounds__(256)
void cvt_x_kernel(const f32x4* __restrict__ x, u32x2* __restrict__ xb) {
    const int idx = blockIdx.x * 256 + threadIdx.x;
    const f32x4 v = x[idx];
    u32x2 p;
    p.x = f32_to_bf16_bits(v.x) | (f32_to_bf16_bits(v.y) << 16);
    p.y = f32_to_bf16_bits(v.z) | (f32_to_bf16_bits(v.w) << 16);
    xb[idx] = p;
}

// ---------------------------------------------------------------------------
// Kernel 3: bf16 WMMA GEMM with async global->LDS double-buffered pipeline
// ---------------------------------------------------------------------------
#define BM 128
#define BN 128
#define BK 64
#define LDSP 72   // padded LDS row stride (144B, keeps 16B alignment)

union FragAB { i32x4 q[2]; v16bf v; };

// Issue 8 async 16B global->LDS copies for this thread's share of an
// (A,B) 128x64 bf16 tile pair. GVS mode: SGPR64 base + VGPR32 byte offset.
__device__ __forceinline__ void async_tile_load(
    unsigned int ldsA, unsigned int ldsB,
    const unsigned short* __restrict__ Xb,
    const unsigned short* __restrict__ Wb,
    int m0, int n0, int k0, int tid) {
#pragma unroll
    for (int t = 0; t < 4; ++t) {
        const int chunk = tid + t * 256;
        const int row = chunk >> 3;           // 0..127
        const int c8  = (chunk & 7) << 3;     // element offset 0..56
        const unsigned int lofs = (unsigned int)(row * LDSP + c8) * 2u;
        const unsigned int ga = (unsigned int)((m0 + row) * IN_K + k0 + c8) * 2u;
        const unsigned int gb = (unsigned int)((n0 + row) * IN_K + k0 + c8) * 2u;
        asm volatile("global_load_async_to_lds_b128 %0, %1, %2"
                     :: "v"(ldsA + lofs), "v"(ga), "s"(Xb) : "memory");
        asm volatile("global_load_async_to_lds_b128 %0, %1, %2"
                     :: "v"(ldsB + lofs), "v"(gb), "s"(Wb) : "memory");
    }
}

__global__ __launch_bounds__(256)
void nf4_gemm_wmma_kernel(const unsigned short* __restrict__ Xb,
                          const unsigned short* __restrict__ Wb,
                          const float* __restrict__ bias,
                          float* __restrict__ out) {
    __shared__ __align__(16) unsigned short As[2][BM * LDSP];
    __shared__ __align__(16) unsigned short Bs[2][BN * LDSP];

    const int tid  = threadIdx.x;
    const int lane = tid & 31;
    const int wave = tid >> 5;
    const int wm   = wave >> 1;          // 0..3 : 32-row strip of M
    const int wn   = wave & 1;           // 0..1 : 64-col strip of N
    const int m0   = blockIdx.y * BM;
    const int n0   = blockIdx.x * BN;

    const int lhalf = lane >> 4;
    const int l15   = lane & 15;

    // LDS base byte offsets (low 32 bits of generic shared address)
    const unsigned int aBase0 = (unsigned int)(size_t)(&As[0][0]);
    const unsigned int aBase1 = (unsigned int)(size_t)(&As[1][0]);
    const unsigned int bBase0 = (unsigned int)(size_t)(&Bs[0][0]);
    const unsigned int bBase1 = (unsigned int)(size_t)(&Bs[1][0]);

    v8f acc[2][4];
#pragma unroll
    for (int mi = 0; mi < 2; ++mi)
#pragma unroll
        for (int ni = 0; ni < 4; ++ni)
            acc[mi][ni] = (v8f)0.0f;

    const int NIT = IN_K / BK;           // 64

    // Prologue: fetch tile 0 into buffer 0
    async_tile_load(aBase0, bBase0, Xb, Wb, m0, n0, 0, tid);

    for (int i = 0; i < NIT; ++i) {
        const int buf = i & 1;
        // Prefetch tile i+1 into the other buffer (safe: everyone finished
        // reading it at the trailing barrier of iteration i-1).
        if (i + 1 < NIT) {
            async_tile_load(buf ? aBase0 : aBase1, buf ? bBase0 : bBase1,
                            Xb, Wb, m0, n0, (i + 1) * BK, tid);
            // async loads complete in order: <=8 outstanding means tile i's
            // 8 copies have landed; the prefetch may still be in flight.
            asm volatile("s_wait_asynccnt 0x8" ::: "memory");
        } else {
            asm volatile("s_wait_asynccnt 0x0" ::: "memory");
        }
        __syncthreads();

        const unsigned short* __restrict__ At = &As[buf][0];
        const unsigned short* __restrict__ Bt = &Bs[buf][0];

#pragma unroll
        for (int ks = 0; ks < 2; ++ks) {
            const int kk = ks * 32;
            // A frag (16x32): lanes 0-15 K{0..7,16..23}, lanes 16-31 K{8..15,24..31}
            FragAB a[2];
#pragma unroll
            for (int mi = 0; mi < 2; ++mi) {
                const int r  = wm * 32 + mi * 16 + l15;
                const int cb = kk + lhalf * 8;
                a[mi].q[0] = *(const i32x4*)&At[r * LDSP + cb];
                a[mi].q[1] = *(const i32x4*)&At[r * LDSP + cb + 16];
            }
#pragma unroll
            for (int ni = 0; ni < 4; ++ni) {
                // B frag (32x16): lane = N col; lanes 0-15 K{0..7,16..23}, 16-31 K{8..15,24..31}
                FragAB b;
                const int r  = wn * 64 + ni * 16 + l15;
                const int cb = kk + lhalf * 8;
                b.q[0] = *(const i32x4*)&Bt[r * LDSP + cb];
                b.q[1] = *(const i32x4*)&Bt[r * LDSP + cb + 16];
#pragma unroll
                for (int mi = 0; mi < 2; ++mi) {
                    acc[mi][ni] = __builtin_amdgcn_wmma_f32_16x16x32_bf16(
                        false, a[mi].v, false, b.v,
                        (short)0, acc[mi][ni], false, false);
                }
            }
        }
        __syncthreads();   // protect buf from next iteration's prefetch
    }

    // Epilogue: C/D layout — VGPR r: M = r + 8*lhalf, N = lane&15
#pragma unroll
    for (int ni = 0; ni < 4; ++ni) {
        const int n  = n0 + wn * 64 + ni * 16 + l15;
        const float bv = bias[n];
#pragma unroll
        for (int mi = 0; mi < 2; ++mi) {
            const int mb = m0 + wm * 32 + mi * 16 + lhalf * 8;
#pragma unroll
            for (int r = 0; r < 8; ++r) {
                out[(size_t)(mb + r) * OUT_N + n] = acc[mi][ni][r] + bv;
            }
        }
    }
}

// ---------------------------------------------------------------------------
extern "C" void kernel_launch(void* const* d_in, const int* in_sizes, int n_in,
                              void* d_out, int out_size, void* d_ws, size_t ws_size,
                              hipStream_t stream) {
    const float* x     = (const float*)d_in[0];   // (2,2048,4096) f32
    const int*   wq    = (const int*)  d_in[1];   // (11008,2048) int32 bytes
    const float* scale = (const float*)d_in[2];   // (11008,)
    const float* bias  = (const float*)d_in[3];   // (11008,)
    float* out = (float*)d_out;                   // (2,2048,11008) f32

    unsigned short* Wb = (unsigned short*)d_ws;                 // N*K bf16
    unsigned short* Xb = Wb + (size_t)OUT_N * IN_K;             // M*K bf16

    nf4_dequant_kernel<<<(OUT_N * QCOLS) / 256, 256, 0, stream>>>(
        wq, scale, (unsigned int*)Wb);

    cvt_x_kernel<<<(TOK_M * IN_K) / (4 * 256), 256, 0, stream>>>(
        (const f32x4*)x, (u32x2*)Xb);

    dim3 grid(OUT_N / BN, TOK_M / BM);
    nf4_gemm_wmma_kernel<<<grid, 256, 0, stream>>>(Xb, Wb, bias, out);
}
